// TransformerDecoderWithCrossAttention_83597243449967
// MI455X (gfx1250) — matrix operand order repaired and checked by reading, not compiled, not run on hardware
//
#include <hip/hip_runtime.h>
#include <hip/hip_bf16.h>

// ---------------------------------------------------------------------------
// MI455X (gfx1250) transformer decoder forward, bf16 WMMA everywhere.
// Round 4: weights pre-transposed at conversion time so BOTH GEMM operands
// stage as contiguous async b128 copies; double-buffered LDS pipelines
// (issue stage t+1, s_wait_asynccnt(2), compute stage t) in GEMM and
// attention; V projection written transposed so attention V staging is
// contiguous async too.
// ---------------------------------------------------------------------------

typedef __bf16 bf16;
typedef __attribute__((ext_vector_type(16))) __bf16 v16bf;
typedef __attribute__((ext_vector_type(8)))  float  v8f;
typedef __attribute__((ext_vector_type(4)))  int    v4i;
typedef __attribute__((ext_vector_type(2)))  int    v2i;

typedef __attribute__((address_space(1))) v4i gv4i;   // global-AS 16B vector
typedef __attribute__((address_space(3))) v4i lv4i;   // LDS-AS 16B vector

constexpr int B_ = 32, T_ = 512, S_ = 256, ED_ = 768;
constexpr int D_ = 256, H_ = 8, L_ = 4, FF_ = 1024, HD_ = 32;
constexpr float NEG_INF = -1e9f;

// ---- gfx1250 async global->LDS copy (fallback: plain 16B copy) -------------
#if defined(__gfx1250__) && __has_builtin(__builtin_amdgcn_global_load_async_to_lds_b128) && __has_builtin(__builtin_amdgcn_s_wait_asynccnt)
#define USE_ASYNC_LDS 1
#else
#define USE_ASYNC_LDS 0
#endif

__device__ __forceinline__ void copy16_g2l(bf16* lds, const bf16* g) {
#if USE_ASYNC_LDS
    __builtin_amdgcn_global_load_async_to_lds_b128((gv4i*)g, (lv4i*)lds, 0, 0);
#else
    *(v4i*)lds = *(const v4i*)g;
#endif
}
template <int N>
__device__ __forceinline__ void async_join_n() {
#if USE_ASYNC_LDS
    __builtin_amdgcn_s_wait_asynccnt(N);
#endif
}

// ---- fragment assembly: two aligned 16B LDS chunks -> v16bf ----------------
struct V16Parts { v4i lo, hi; };
__device__ __forceinline__ v16bf make_frag(const bf16* p0, const bf16* p1) {
    V16Parts p{*(const v4i*)p0, *(const v4i*)p1};
    return __builtin_bit_cast(v16bf, p);
}

// ---------------------------------------------------------------------------
// f32 -> bf16 conversion, 4 elements/thread
// ---------------------------------------------------------------------------
__global__ void f32_to_bf16_kernel(const float* __restrict__ in, bf16* __restrict__ out, int n4) {
    int i = blockIdx.x * blockDim.x + threadIdx.x;
    int stride = gridDim.x * blockDim.x;
    for (; i < n4; i += stride) {
        const float4 v = *(const float4*)(in + (size_t)i * 4);
        bf16 ob[4] __attribute__((aligned(8)));
        ob[0] = (bf16)v.x; ob[1] = (bf16)v.y; ob[2] = (bf16)v.z; ob[3] = (bf16)v.w;
        *(v2i*)(out + (size_t)i * 4) = *(const v2i*)ob;
    }
}

// ---------------------------------------------------------------------------
// f32[K][N] -> bf16 TRANSPOSED out[N][K]; coalesced reads (lane-major n),
// one 16B store per thread. One-time pass per weight.
// ---------------------------------------------------------------------------
__global__ void f32_to_bf16_t_kernel(const float* __restrict__ in, bf16* __restrict__ out,
                                     int K, int N) {
    int i = blockIdx.x * blockDim.x + threadIdx.x;   // over N * K/8
    if (i >= N * (K >> 3)) return;
    const int n  = i % N;
    const int kc = (i / N) << 3;
    bf16 ob[8] __attribute__((aligned(16)));
#pragma unroll
    for (int j = 0; j < 8; ++j) ob[j] = (bf16)in[(size_t)(kc + j) * N + n];
    *(v4i*)(out + (size_t)n * K + kc) = *(const v4i*)ob;
}

// ---------------------------------------------------------------------------
// Embedding + sinusoidal positional encoding, 4 elements/thread
// ---------------------------------------------------------------------------
__global__ void embed_kernel(const float* __restrict__ tgt, const float* __restrict__ W_emb,
                             const float* __restrict__ b_emb,
                             float* __restrict__ xf, bf16* __restrict__ xb) {
    int i = blockIdx.x * blockDim.x + threadIdx.x;   // over B*T*D/4
    if (i >= B_ * T_ * D_ / 4) return;
    const int d0 = (i * 4) % D_;
    const int bt = (i * 4) / D_;
    const int t  = bt % T_;
    const float tv = tgt[bt];
    float4 o;
    float* op = (float*)&o;
    bf16 ob[4] __attribute__((aligned(8)));
#pragma unroll
    for (int j = 0; j < 4; ++j) {
        const int d = d0 + j;
        float v = tv * W_emb[d] + b_emb[d];
        const int ii = d >> 1;
        const float div = __expf(-__logf(10000.f) * (2.f * (float)ii) / (float)D_);
        const float ang = (float)t * div;
        v += (d & 1) ? __cosf(ang) : __sinf(ang);
        op[j] = v; ob[j] = (bf16)v;
    }
    *(float4*)(xf + (size_t)i * 4) = o;
    *(v2i*)(xb + (size_t)i * 4)    = *(const v2i*)ob;
}

// ---------------------------------------------------------------------------
// WMMA GEMM: out = epi(A[MxK](bf16) @ W + bias) (+res), W given TRANSPOSED
// as WT[N][K] bf16. Tile 64x64, BK=32, 256 threads = 8 waves (16x32/wave).
// Double-buffered LDS: async-issue stage t+1, s_wait_asynccnt(2), compute t.
// EPI: 0 = bias, 1 = bias+GELU(exact), 2 = bias+residual
// OUTT: write bf16 output transposed (out[col*M + row]) with 16B row-packs.
// ---------------------------------------------------------------------------
template <int EPI, bool OUTB, bool OUTF, bool OUTT>
__global__ void gemm_wmma_bf16(const bf16* __restrict__ A, const bf16* __restrict__ WT,
                               const float* __restrict__ bias, const float* __restrict__ res,
                               bf16* __restrict__ outb, float* __restrict__ outf,
                               int M, int N, int K) {
    __shared__ bf16 As[2][64][40];     // 80B rows: every 16B chunk aligned
    __shared__ bf16 BsT[2][64][40];    // [n][k]

    const int tid  = threadIdx.x;
    const int lane = tid & 31;
    const int w    = tid >> 5;
    const int wr   = (w >> 1) << 4;
    const int wc   = (w & 1) << 5;
    const int m0   = blockIdx.x << 6;
    const int n0   = blockIdx.y << 6;

    v8f acc0 = {};
    v8f acc1 = {};

    const int srow = tid >> 2;          // staging: 64 rows x 32 k, 16B/thread
    const int scol = (tid & 3) << 3;

    const int mrow = wr + (lane & 15);
    const int kb   = (lane >> 4) << 3;  // 0 / 8
    const int ncol = wc + (lane & 15);
    const int kh   = (lane >> 4) << 4;  // 0 / 16

    const bf16* aRow = A  + (size_t)(m0 + srow) * K + scol;
    const bf16* wRow = WT + (size_t)(n0 + srow) * K + scol;

    auto stage = [&](int k0, int buf) {
        copy16_g2l(&As[buf][srow][scol],  aRow + k0);
        copy16_g2l(&BsT[buf][srow][scol], wRow + k0);
    };

    const int nk = K >> 5;
    stage(0, 0);
    for (int it = 0; it < nk; ++it) {
        const int cur = it & 1;
        if (it + 1 < nk) {
            stage((it + 1) << 5, cur ^ 1);
            if (it + 2 < nk) {   // global_prefetch_b8 two tiles ahead
                __builtin_prefetch(aRow + ((it + 2) << 5), 0, 1);
                __builtin_prefetch(wRow + ((it + 2) << 5), 0, 1);
            }
            async_join_n<2>();   // stage `it` landed; stage `it+1` in flight
        } else {
            async_join_n<0>();
        }
        __syncthreads();

        const v16bf a  = make_frag(&As[cur][mrow][kb], &As[cur][mrow][16 + kb]);
        const v16bf b0 = make_frag(&BsT[cur][ncol][kb], &BsT[cur][ncol][16 + kb]);
        const v16bf b1 = make_frag(&BsT[cur][ncol + 16][kb], &BsT[cur][ncol + 16][16 + kb]);
        acc0 = __builtin_amdgcn_wmma_f32_16x16x32_bf16(false, a, false, b0, (short)0, acc0, false, false);
        acc1 = __builtin_amdgcn_wmma_f32_16x16x32_bf16(false, a, false, b1, (short)0, acc1, false, false);
        __syncthreads();
    }

    // Epilogue: lanes 0-15: M=r, lanes 16-31: M=8+r; N = lane&15
    const int colt  = lane & 15;
    const int rhalf = (lane >> 4) << 3;
    if constexpr (OUTT) {
        const int r0 = m0 + wr + rhalf;      // 8 consecutive rows per lane
#pragma unroll
        for (int cth = 0; cth < 2; ++cth) {
            const int col = n0 + wc + (cth << 4) + colt;
            const float bv = bias[col];
            bf16 ob[8] __attribute__((aligned(16)));
#pragma unroll
            for (int r = 0; r < 8; ++r)
                ob[r] = (bf16)(((cth == 0) ? acc0[r] : acc1[r]) + bv);
            *(v4i*)(outb + (size_t)col * M + r0) = *(const v4i*)ob;
        }
    } else {
#pragma unroll
        for (int r = 0; r < 8; ++r) {
            const int row = m0 + wr + rhalf + r;
#pragma unroll
            for (int cth = 0; cth < 2; ++cth) {
                const int col = n0 + wc + (cth << 4) + colt;
                float v = (cth == 0) ? acc0[r] : acc1[r];
                v += bias[col];
                if constexpr (EPI == 1) v = 0.5f * v * (1.f + erff(v * 0.70710678118f));
                if constexpr (EPI == 2) v += res[(size_t)row * N + col];
                if constexpr (OUTF) outf[(size_t)row * N + col] = v;
                if constexpr (OUTB) outb[(size_t)row * N + col] = (bf16)v;
            }
        }
    }
}

// ---------------------------------------------------------------------------
// Fused attention, one (b, h, 16-query-row tile) per 64-thread block (2 waves).
// K tiles row-major (scores B-frag = K^T is contiguous); V comes PRE-
// TRANSPOSED from the V projection (VTg[d][row]) so V staging is contiguous
// async b128 too. Both loops double-buffered on a shared KV tile.
// ---------------------------------------------------------------------------
template <bool CAUSAL>
__global__ void attn_wmma(const bf16* __restrict__ Q, const bf16* __restrict__ Kg,
                          const bf16* __restrict__ VTg, bf16* __restrict__ O,
                          int Ts, int Mrows, float scale) {
    __shared__ bf16  Qs[16][40];
    __shared__ float Sc[16][520];
    __shared__ bf16  Ps[16][512];
    __shared__ bf16  KVs[2][32][40];   // K phase: [s][hd]; V phase: [hd][s]
    __shared__ float red[16][4];

    const int tid  = threadIdx.x;
    const int lane = tid & 31;
    const int w    = tid >> 5;      // 0..1
    const int q0   = blockIdx.x << 4;
    const int h    = blockIdx.y;
    const int b    = blockIdx.z;

    const size_t qbase = ((size_t)b * T_) * D_ + (size_t)h * HD_;
    const size_t kbase = ((size_t)b * Ts) * D_ + (size_t)h * HD_;

    // Stage Q tile (16x32), join, build invariant A-fragment
    {
        const int r = tid >> 2, c = (tid & 3) << 3;
        copy16_g2l(&Qs[r][c], Q + qbase + (size_t)(q0 + r) * D_ + c);
    }
    async_join_n<0>();
    __syncthreads();

    const int mrow = lane & 15;
    const int kb   = (lane >> 4) << 3;
    const int n    = lane & 15;
    const int kh   = (lane >> 4) << 4;
    const v16bf aq = make_frag(&Qs[mrow][kb], &Qs[mrow][16 + kb]);

    const int slen  = CAUSAL ? (q0 + 16) : Ts;   // fully-masked tiles skipped
    const int niter = (slen + 31) >> 5;

    // K staging: 32 s-rows x 32 hd, two 16B async chunks per thread
    const int kr = tid >> 1, kc = (tid & 1) << 4;
    auto stageK = [&](int ks, int buf) {
        const bf16* kp = Kg + kbase + (size_t)(ks + kr) * D_ + kc;
        copy16_g2l(&KVs[buf][kr][kc], kp);
        copy16_g2l(&KVs[buf][kr][kc + 8], kp + 8);
    };
    // V staging (transposed source): 32 hd-rows x 32 s, contiguous async
    const int vhd = tid & 31, vsg = (tid >> 5) << 4;
    const size_t vbase = (size_t)(h * HD_ + vhd) * Mrows + (size_t)b * Ts;
    auto stageV = [&](int ks, int buf) {
        const bf16* vp = VTg + vbase + ks + vsg;
        copy16_g2l(&KVs[buf][vhd][vsg], vp);
        copy16_g2l(&KVs[buf][vhd][vsg + 8], vp + 8);
    };

    // ---- scores = scale * Q K^T (+causal) -> Sc   (double-buffered)
    stageK(0, 0);
    for (int it = 0; it < niter; ++it) {
        const int cur = it & 1;
        if (it + 1 < niter) { stageK((it + 1) << 5, cur ^ 1); async_join_n<2>(); }
        else                { async_join_n<0>(); }
        __syncthreads();
        const int sb = (it << 5) + (w << 4);
        const v16bf bk = make_frag(&KVs[cur][(w << 4) + n][kb], &KVs[cur][(w << 4) + n][16 + kb]);
        v8f cacc = {};
        cacc = __builtin_amdgcn_wmma_f32_16x16x32_bf16(false, aq, false, bk, (short)0, cacc, false, false);
        const int rhalf = (lane >> 4) << 3;
#pragma unroll
        for (int r = 0; r < 8; ++r) {
            const int row = rhalf + r;
            const int s   = sb + n;
            float v = cacc[r] * scale;
            if (CAUSAL && s > q0 + row) v = NEG_INF;
            Sc[row][s] = v;
        }
        __syncthreads();
    }

    // ---- softmax (4 partials/row, 4-wide contiguous chunks) -> Ps (bf16)
    {
        const int row = tid & 15, part = tid >> 4;
        float m = -3.4e38f;
        for (int c0 = part << 2; c0 < slen; c0 += 16) {
#pragma unroll
            for (int j = 0; j < 4; ++j) m = fmaxf(m, Sc[row][c0 + j]);
        }
        red[row][part] = m;
        __syncthreads();
        const float rm = fmaxf(fmaxf(red[row][0], red[row][1]), fmaxf(red[row][2], red[row][3]));
        __syncthreads();
        float ssum = 0.f;
        for (int c0 = part << 2; c0 < slen; c0 += 16) {
#pragma unroll
            for (int j = 0; j < 4; ++j) {
                const float e = __expf(Sc[row][c0 + j] - rm);
                Sc[row][c0 + j] = e;
                ssum += e;
            }
        }
        red[row][part] = ssum;
        __syncthreads();
        const float rs  = red[row][0] + red[row][1] + red[row][2] + red[row][3];
        const float inv = 1.f / rs;
        for (int c0 = part << 2; c0 < slen; c0 += 16) {
            bf16 pb[4] __attribute__((aligned(8)));
#pragma unroll
            for (int j = 0; j < 4; ++j) pb[j] = (bf16)(Sc[row][c0 + j] * inv);
            *(v2i*)&Ps[row][c0] = *(const v2i*)pb;
        }
        const int slenp = niter << 5;           // pad to k-step with zeros
        for (int c0 = slen + (part << 2); c0 < slenp; c0 += 16) {
            v2i z = {};
            *(v2i*)&Ps[row][c0] = z;
        }
    }
    __syncthreads();

    // ---- O = P @ V (wave w owns 16 output hd-cols)   (double-buffered)
    v8f acco = {};
    stageV(0, 0);
    for (int it = 0; it < niter; ++it) {
        const int cur = it & 1;
        if (it + 1 < niter) { stageV((it + 1) << 5, cur ^ 1); async_join_n<2>(); }
        else                { async_join_n<0>(); }
        __syncthreads();
        const int ks = it << 5;
        const v16bf ap = make_frag(&Ps[mrow][ks + kb], &Ps[mrow][ks + 16 + kb]);
        const v16bf bv = make_frag(&KVs[cur][(w << 4) + n][kb], &KVs[cur][(w << 4) + n][16 + kb]);
        acco = __builtin_amdgcn_wmma_f32_16x16x32_bf16(false, ap, false, bv, (short)0, acco, false, false);
        __syncthreads();
    }

    const int rhalf = (lane >> 4) << 3;
#pragma unroll
    for (int r = 0; r < 8; ++r) {
        const int row = q0 + rhalf + r;
        const int col = h * HD_ + (w << 4) + n;
        O[((size_t)b * T_ + row) * D_ + col] = (bf16)acco[r];
    }
}

// ---------------------------------------------------------------------------
// LayerNorm over D=256: one block (64 threads) per row, one float4 per thread
// ---------------------------------------------------------------------------
__global__ void ln_kernel(const float* __restrict__ in, const float* __restrict__ g,
                          const float* __restrict__ be,
                          float* __restrict__ xf, bf16* __restrict__ xb) {
    const int row = blockIdx.x;
    const int tid = threadIdx.x;
    __shared__ float sred[64];

    const float4 v4 = *(const float4*)(in + (size_t)row * D_ + tid * 4);
    sred[tid] = v4.x + v4.y + v4.z + v4.w;
    __syncthreads();
    for (int off = 32; off > 0; off >>= 1) {
        if (tid < off) sred[tid] += sred[tid + off];
        __syncthreads();
    }
    const float mean = sred[0] / (float)D_;
    __syncthreads();

    const float dx = v4.x - mean, dy = v4.y - mean, dz = v4.z - mean, dw = v4.w - mean;
    sred[tid] = dx * dx + dy * dy + dz * dz + dw * dw;
    __syncthreads();
    for (int off = 32; off > 0; off >>= 1) {
        if (tid < off) sred[tid] += sred[tid + off];
        __syncthreads();
    }
    const float inv = rsqrtf(sred[0] / (float)D_ + 1e-5f);

    float4 o;
    o.x = dx * inv * g[tid * 4 + 0] + be[tid * 4 + 0];
    o.y = dy * inv * g[tid * 4 + 1] + be[tid * 4 + 1];
    o.z = dz * inv * g[tid * 4 + 2] + be[tid * 4 + 2];
    o.w = dw * inv * g[tid * 4 + 3] + be[tid * 4 + 3];
    *(float4*)(xf + (size_t)row * D_ + tid * 4) = o;
    bf16 ob[4] __attribute__((aligned(8)));
    ob[0] = (bf16)o.x; ob[1] = (bf16)o.y; ob[2] = (bf16)o.z; ob[3] = (bf16)o.w;
    *(v2i*)(xb + (size_t)row * D_ + tid * 4) = *(const v2i*)ob;
}

// ---------------------------------------------------------------------------
// Final head: (B*T,128) bf16 @ (128,1) f32 + bias -> f32
// ---------------------------------------------------------------------------
__global__ void head2_kernel(const bf16* __restrict__ h1, const float* __restrict__ Wp2,
                             const float* __restrict__ bp2, float* __restrict__ out) {
    int i = blockIdx.x * blockDim.x + threadIdx.x;
    if (i >= B_ * T_) return;
    const v4i* p4 = (const v4i*)(h1 + (size_t)i * (D_ / 2));
    float s = bp2[0];
#pragma unroll
    for (int kk = 0; kk < 16; ++kk) {
        bf16 tb[8] __attribute__((aligned(16)));
        *(v4i*)tb = p4[kk];
#pragma unroll
        for (int j = 0; j < 8; ++j) s += (float)tb[j] * Wp2[kk * 8 + j];
    }
    out[i] = s;
}

// ---------------------------------------------------------------------------
// Host orchestration
// ---------------------------------------------------------------------------
extern "C" void kernel_launch(void* const* d_in, const int* in_sizes, int n_in,
                              void* d_out, int out_size, void* d_ws, size_t ws_size,
                              hipStream_t stream) {
    (void)in_sizes; (void)n_in; (void)out_size; (void)ws_size;

    const float* tgt    = (const float*)d_in[0];
    const float* memory = (const float*)d_in[1];
    const float* W_emb  = (const float*)d_in[2];
    const float* b_emb  = (const float*)d_in[3];
    const float* W_enc  = (const float*)d_in[4];
    const float* b_enc  = (const float*)d_in[5];
    const float* Wq_s = (const float*)d_in[6];  const float* bq_s = (const float*)d_in[7];
    const float* Wk_s = (const float*)d_in[8];  const float* bk_s = (const float*)d_in[9];
    const float* Wv_s = (const float*)d_in[10]; const float* bv_s = (const float*)d_in[11];
    const float* Wo_s = (const float*)d_in[12]; const float* bo_s = (const float*)d_in[13];
    const float* Wq_c = (const float*)d_in[14]; const float* bq_c = (const float*)d_in[15];
    const float* Wk_c = (const float*)d_in[16]; const float* bk_c = (const float*)d_in[17];
    const float* Wv_c = (const float*)d_in[18]; const float* bv_c = (const float*)d_in[19];
    const float* Wo_c = (const float*)d_in[20]; const float* bo_c = (const float*)d_in[21];
    const float* g1  = (const float*)d_in[22];  const float* be1 = (const float*)d_in[23];
    const float* g2  = (const float*)d_in[24];  const float* be2 = (const float*)d_in[25];
    const float* g3  = (const float*)d_in[26];  const float* be3 = (const float*)d_in[27];
    const float* Wf1 = (const float*)d_in[28];  const float* bf1 = (const float*)d_in[29];
    const float* Wf2 = (const float*)d_in[30];  const float* bf2 = (const float*)d_in[31];
    const float* Wp1 = (const float*)d_in[32];  const float* bp1 = (const float*)d_in[33];
    const float* Wp2 = (const float*)d_in[34];  const float* bp2 = (const float*)d_in[35];

    char* ws = (char*)d_ws;
    size_t off = 0;
    auto alloc = [&](size_t bytes) -> void* {
        void* p = ws + off;
        off += (bytes + 255) & ~(size_t)255;
        return p;
    };

    const int MT = B_ * T_;   // 16384
    const int MS = B_ * S_;   // 8192

    float* xf    = (float*)alloc((size_t)MT * D_ * 4);
    bf16*  xb    = (bf16*) alloc((size_t)MT * D_ * 2);
    float* tmp   = (float*)alloc((size_t)MT * D_ * 4);
    bf16*  attnb = (bf16*) alloc((size_t)MT * D_ * 2);
    bf16*  qb    = (bf16*) alloc((size_t)MT * D_ * 2);
    bf16*  kbuf  = (bf16*) alloc((size_t)MT * D_ * 2);
    bf16*  vbufT = (bf16*) alloc((size_t)MT * D_ * 2);   // transposed V [d][row]
    bf16*  memin = (bf16*) alloc((size_t)MS * ED_ * 2);
    bf16*  memb  = (bf16*) alloc((size_t)MS * D_ * 2);
    bf16*  hid   = (bf16*) alloc((size_t)MT * FF_ * 2);
    bf16*  h1    = (bf16*) alloc((size_t)MT * (D_ / 2) * 2);

    bf16* wencT = (bf16*)alloc((size_t)ED_ * D_ * 2);
    bf16* wp1T  = (bf16*)alloc((size_t)D_ * (D_ / 2) * 2);
    bf16 *wqs[L_], *wks[L_], *wvs[L_], *wos[L_], *wqc[L_], *wkc[L_], *wvc[L_], *woc[L_], *wf1T[L_], *wf2T[L_];
    for (int l = 0; l < L_; ++l) {
        wqs[l]  = (bf16*)alloc((size_t)D_ * D_ * 2);
        wks[l]  = (bf16*)alloc((size_t)D_ * D_ * 2);
        wvs[l]  = (bf16*)alloc((size_t)D_ * D_ * 2);
        wos[l]  = (bf16*)alloc((size_t)D_ * D_ * 2);
        wqc[l]  = (bf16*)alloc((size_t)D_ * D_ * 2);
        wkc[l]  = (bf16*)alloc((size_t)D_ * D_ * 2);
        wvc[l]  = (bf16*)alloc((size_t)D_ * D_ * 2);
        woc[l]  = (bf16*)alloc((size_t)D_ * D_ * 2);
        wf1T[l] = (bf16*)alloc((size_t)D_ * FF_ * 2);
        wf2T[l] = (bf16*)alloc((size_t)FF_ * D_ * 2);
    }

    auto conv = [&](const float* src, bf16* dst, int n) {
        const int n4 = n / 4;
        int blocks = (n4 + 255) / 256;
        if (blocks > 4096) blocks = 4096;
        f32_to_bf16_kernel<<<blocks, 256, 0, stream>>>(src, dst, n4);
    };
    auto convT = [&](const float* src, bf16* dst, int K, int N) {   // W[K][N] -> WT[N][K]
        const int total = N * (K / 8);
        f32_to_bf16_t_kernel<<<(total + 255) / 256, 256, 0, stream>>>(src, dst, K, N);
    };

    // ---- precision conversion + weight transposition (one-time, L2 resident)
    conv(memory, memin, MS * ED_);
    convT(W_enc, wencT, ED_, D_);
    convT(Wp1, wp1T, D_, D_ / 2);
    for (int l = 0; l < L_; ++l) {
        convT(Wq_s + (size_t)l * D_ * D_, wqs[l], D_, D_);
        convT(Wk_s + (size_t)l * D_ * D_, wks[l], D_, D_);
        convT(Wv_s + (size_t)l * D_ * D_, wvs[l], D_, D_);
        convT(Wo_s + (size_t)l * D_ * D_, wos[l], D_, D_);
        convT(Wq_c + (size_t)l * D_ * D_, wqc[l], D_, D_);
        convT(Wk_c + (size_t)l * D_ * D_, wkc[l], D_, D_);
        convT(Wv_c + (size_t)l * D_ * D_, wvc[l], D_, D_);
        convT(Wo_c + (size_t)l * D_ * D_, woc[l], D_, D_);
        convT(Wf1 + (size_t)l * D_ * FF_, wf1T[l], D_, FF_);
        convT(Wf2 + (size_t)l * FF_ * D_, wf2T[l], FF_, D_);
    }

    // ---- embedding + positional encoding
    embed_kernel<<<(MT * D_ / 4 + 255) / 256, 256, 0, stream>>>(tgt, W_emb, b_emb, xf, xb);

    // ---- encoder memory projection
    {
        dim3 g(MS / 64, D_ / 64);
        gemm_wmma_bf16<0, true, false, false><<<g, 256, 0, stream>>>(memin, wencT, b_enc, nullptr,
                                                                     memb, nullptr, MS, D_, ED_);
    }

    const dim3 gT(MT / 64, D_ / 64);
    const dim3 gS(MS / 64, D_ / 64);
    const dim3 gA(T_ / 16, H_, B_);
    const float scale = 0.17677669529663687f;   // 1/sqrt(32)

    for (int l = 0; l < L_; ++l) {
        // ---- self-attention
        gemm_wmma_bf16<0, true, false, false><<<gT, 256, 0, stream>>>(xb, wqs[l], bq_s + (size_t)l * D_, nullptr, qb,    nullptr, MT, D_, D_);
        gemm_wmma_bf16<0, true, false, false><<<gT, 256, 0, stream>>>(xb, wks[l], bk_s + (size_t)l * D_, nullptr, kbuf,  nullptr, MT, D_, D_);
        gemm_wmma_bf16<0, false, false, true><<<gT, 256, 0, stream>>>(xb, wvs[l], bv_s + (size_t)l * D_, nullptr, vbufT, nullptr, MT, D_, D_);
        attn_wmma<true><<<gA, 64, 0, stream>>>(qb, kbuf, vbufT, attnb, T_, MT, scale);
        gemm_wmma_bf16<2, false, true, false><<<gT, 256, 0, stream>>>(attnb, wos[l], bo_s + (size_t)l * D_, xf, nullptr, tmp, MT, D_, D_);
        ln_kernel<<<MT, 64, 0, stream>>>(tmp, g1 + (size_t)l * D_, be1 + (size_t)l * D_, xf, xb);

        // ---- cross-attention
        gemm_wmma_bf16<0, true, false, false><<<gT, 256, 0, stream>>>(xb,   wqc[l], bq_c + (size_t)l * D_, nullptr, qb,    nullptr, MT, D_, D_);
        gemm_wmma_bf16<0, true, false, false><<<gS, 256, 0, stream>>>(memb, wkc[l], bk_c + (size_t)l * D_, nullptr, kbuf,  nullptr, MS, D_, D_);
        gemm_wmma_bf16<0, false, false, true><<<gS, 256, 0, stream>>>(memb, wvc[l], bv_c + (size_t)l * D_, nullptr, vbufT, nullptr, MS, D_, D_);
        attn_wmma<false><<<gA, 64, 0, stream>>>(qb, kbuf, vbufT, attnb, S_, MS, scale);
        gemm_wmma_bf16<2, false, true, false><<<gT, 256, 0, stream>>>(attnb, woc[l], bo_c + (size_t)l * D_, xf, nullptr, tmp, MT, D_, D_);
        ln_kernel<<<MT, 64, 0, stream>>>(tmp, g2 + (size_t)l * D_, be2 + (size_t)l * D_, xf, xb);

        // ---- feed-forward
        {
            dim3 gF(MT / 64, FF_ / 64);
            gemm_wmma_bf16<1, true, false, false><<<gF, 256, 0, stream>>>(xb, wf1T[l], bf1 + (size_t)l * FF_, nullptr, hid, nullptr, MT, FF_, D_);
        }
        gemm_wmma_bf16<2, false, true, false><<<gT, 256, 0, stream>>>(hid, wf2T[l], bf2 + (size_t)l * D_, xf, nullptr, tmp, MT, D_, FF_);
        ln_kernel<<<MT, 64, 0, stream>>>(tmp, g3 + (size_t)l * D_, be3 + (size_t)l * D_, xf, xb);
    }

    // ---- prediction head
    {
        dim3 gH(MT / 64, (D_ / 2) / 64);
        gemm_wmma_bf16<1, true, false, false><<<gH, 256, 0, stream>>>(xb, wp1T, bp1, nullptr, h1, nullptr, MT, D_ / 2, D_);
    }
    head2_kernel<<<(MT + 255) / 256, 256, 0, stream>>>(h1, Wp2, bp2, (float*)d_out);
}
